// PNAConv_OnlyNodes_44281112822529
// MI455X (gfx1250) — compile-verified, matrix operand order checked by reading, not run on hardware
//
#include <hip/hip_runtime.h>
#include <math.h>

// ---------------------------------------------------------------------------
// MI455X (gfx1250) PNAConv forward — bf16 WMMA everywhere, wave32.
// Round 3: native vector f32->bf16 converts (__builtin_convertvector) instead
// of manual RNE bit math; fast rcp sigmoid. Same fused structure as round 2.
// ---------------------------------------------------------------------------

typedef __attribute__((ext_vector_type(16))) __bf16 v16bf;
typedef __attribute__((ext_vector_type(8)))  __bf16 v8bf;
typedef __attribute__((ext_vector_type(8)))  float  v8f;
typedef __attribute__((ext_vector_type(4)))  float  f4v;

// 16-bit A/B fragment K-index for lane-half (ISA 7.12.2, 16-bit A 16x32)
__device__ __forceinline__ int kmap(int e, int half) {
  int v = e >> 1, p = e & 1;
  return (v < 4) ? (half * 8 + 2 * v + p) : (16 + half * 8 + 2 * (v - 4) + p);
}

__device__ __forceinline__ v8f wmma_bf16(v16bf a, v16bf b, v8f c) {
  return __builtin_amdgcn_wmma_f32_16x16x32_bf16(false, a, false, b, (short)0, c,
                                                 false, false);
}

__device__ __forceinline__ v16bf cat16(v8bf lo, v8bf hi) {
  return __builtin_shufflevector(lo, hi, 0, 1, 2, 3, 4, 5, 6, 7,
                                 8, 9, 10, 11, 12, 13, 14, 15);
}

__device__ __forceinline__ v8bf cvt8(f4v a, f4v b) {
  v8f t = __builtin_shufflevector(a, b, 0, 1, 2, 3, 4, 5, 6, 7);
  return __builtin_convertvector(t, v8bf);   // fptrunc, RNE
}

__device__ __forceinline__ v16bf mk16(f4v a0, f4v a1, f4v a2, f4v a3) {
  return cat16(cvt8(a0, a1), cvt8(a2, a3));
}

// packed-B fragment fetch: lane's 16 bf16 contiguous -> two b128 loads
__device__ __forceinline__ v16bf ldB(const __bf16* __restrict__ P, int kt, int nj,
                                     int j, int lane) {
  const v8bf* p = (const v8bf*)(P + ((((size_t)kt * nj + j) * 32 + lane) << 4));
  return cat16(p[0], p[1]);
}

// order-preserving float<->uint map for integer atomicMin/Max
__device__ __forceinline__ unsigned fenc(float f) {
  unsigned u = __float_as_uint(f);
  return (u & 0x80000000u) ? ~u : (u | 0x80000000u);
}
__device__ __forceinline__ float fdec(unsigned u) {
  unsigned v = (u & 0x80000000u) ? (u & 0x7FFFFFFFu) : ~u;
  return __uint_as_float(v);
}

// ---------------------------------------------------------------------------
// weight repack: W[K,FO] fp32 -> P bf16 in fragment order
// P[(((kt*NJ)+j)*32 + lane)*16 + e] = W[kt*32+kmap(e,half)][j*16+col] (0 pad)
// ---------------------------------------------------------------------------
__global__ void pack_w(const float* __restrict__ W, __bf16* __restrict__ P,
                       int K, int FO, int KT, int NJ) {
  int i = blockIdx.x * blockDim.x + threadIdx.x;
  int total = KT * NJ * 512;
  if (i >= total) return;
  int e    = i & 15;
  int lane = (i >> 4) & 31;
  int j    = (i >> 9) % NJ;
  int kt   = i / (512 * NJ);
  int half = lane >> 4, col = lane & 15;
  int k = kt * 32 + kmap(e, half);
  int c = j * 16 + col;
  float v = (k < K && c < FO) ? W[(size_t)k * FO + c] : 0.f;
  P[i] = (__bf16)v;
}

// ---------------------------------------------------------------------------
// small setup kernels
// ---------------------------------------------------------------------------
__global__ void fill_f32(float* p, int n, float v) {
  int i = blockIdx.x * blockDim.x + threadIdx.x;
  if (i < n) p[i] = v;
}

__global__ void agg_init_k(float* s, float* s2, unsigned* mn, unsigned* mx, int n) {
  int i = blockIdx.x * blockDim.x + threadIdx.x;
  if (i < n) { s[i] = 0.f; s2[i] = 0.f; mn[i] = 0xFFFFFFFFu; mx[i] = 0u; }
}

__global__ void deg_kernel(const int* __restrict__ ei, int E, float* __restrict__ deg) {
  int e = blockIdx.x * blockDim.x + threadIdx.x;
  if (e < E) atomicAdd(&deg[ei[E + e]], 1.0f);   // dst = row 1 of edge_index
}

__global__ void reduce_deg(const float* __restrict__ deg, int N, float* __restrict__ sums) {
  __shared__ float s0[256], s1[256];
  int i = blockIdx.x * 256 + threadIdx.x;
  float d = (i < N) ? deg[i] : 0.f;
  s0[threadIdx.x] = d;
  s1[threadIdx.x] = (i < N) ? logf(d + 1.f) : 0.f;
  __syncthreads();
  for (int st = 128; st > 0; st >>= 1) {
    if ((int)threadIdx.x < st) {
      s0[threadIdx.x] += s0[threadIdx.x + st];
      s1[threadIdx.x] += s1[threadIdx.x + st];
    }
    __syncthreads();
  }
  if (threadIdx.x == 0) { atomicAdd(&sums[0], s0[0]); atomicAdd(&sums[1], s1[0]); }
}

__global__ void scl_kernel(const float* __restrict__ deg, const float* __restrict__ sums,
                           float* __restrict__ scl, int N) {
  int n = blockIdx.x * blockDim.x + threadIdx.x;
  if (n >= N) return;
  float avg_lin = sums[0] / (float)N;
  float avg_log = sums[1] / (float)N;
  float cc   = fmaxf(deg[n], 1.f);
  float logd = logf(cc + 1.f);
  scl[n * 5 + 0] = 1.f;
  scl[n * 5 + 1] = logd / avg_log;
  scl[n * 5 + 2] = avg_log / logd;
  scl[n * 5 + 3] = cc / avg_lin;
  scl[n * 5 + 4] = avg_lin / cc;
}

// ---------------------------------------------------------------------------
// fused edge-message GEMM + scatter aggregation
// ---------------------------------------------------------------------------
template <int FI>
__global__ void edge_msg_kernel(const float* __restrict__ x,
                                const int* __restrict__ ei, int E,
                                const __bf16* __restrict__ Pw,
                                const float* __restrict__ bpre,
                                float* __restrict__ s, float* __restrict__ s2,
                                unsigned* __restrict__ mn, unsigned* __restrict__ mx) {
  constexpr int K  = 2 * FI;      // 128 or 64
  constexpr int NJ = FI / 16;
  constexpr int NW = 8;
  __shared__ __bf16 ldsX[NW][16 * K];
  __shared__ int    ldsD[NW][16];

  const int w = threadIdx.x >> 5, lane = threadIdx.x & 31;
  const int e0 = (blockIdx.x * NW + w) * 16;
  if (e0 >= E) return;

  // gather 16 edge rows into LDS; one iteration = 8 contiguous columns
  for (int idx = lane; idx < 16 * (K / 8); idx += 32) {
    int t  = idx / (K / 8);
    int c8 = (idx - t * (K / 8)) * 8;
    int e  = e0 + t;
    int row = (c8 < FI) ? ei[E + e] : ei[e];   // dst-half | src-half
    int col = (c8 < FI) ? c8 : c8 - FI;
    const f4v* p = (const f4v*)&x[(size_t)row * FI + col];
    *(v8bf*)&ldsX[w][t * K + c8] = cvt8(p[0], p[1]);
  }
  if (lane < 16) ldsD[w][lane] = ei[E + e0 + lane];
  asm volatile("s_wait_dscnt 0x0" ::: "memory");   // wave-level LDS fence

  const int half = lane >> 4, mrow = lane & 15, col = lane & 15;

  v8f acc[NJ] = {};
#pragma unroll
  for (int kt = 0; kt < K / 32; ++kt) {
    const v8bf* lo = (const v8bf*)&ldsX[w][mrow * K + kt * 32 + half * 8];
    const v8bf* hi = (const v8bf*)&ldsX[w][mrow * K + kt * 32 + 16 + half * 8];
    v16bf av = cat16(lo[0], hi[0]);
#pragma unroll
    for (int j = 0; j < NJ; ++j)
      acc[j] = wmma_bf16(av, ldB(Pw, kt, NJ, j, lane), acc[j]);
  }

  // bias + scatter aggregation (segment sum / sum^2 / min / max by dst)
#pragma unroll
  for (int j = 0; j < NJ; ++j) {
    float bb = bpre[j * 16 + col];
#pragma unroll
    for (int r = 0; r < 8; ++r) {
      float v = acc[j][r] + bb;
      int d  = ldsD[w][r + 8 * half];
      int ai = d * FI + j * 16 + col;
      atomicAdd(&s[ai], v);
      atomicAdd(&s2[ai], v * v);
      atomicMin(&mn[ai], fenc(v));
      atomicMax(&mx[ai], fenc(v));
    }
  }
}

// aggregate finalize: agg5[n] = [sum, mean, min, max, std] per channel
__global__ void agg_fin(const float* __restrict__ s, const float* __restrict__ s2,
                        const unsigned* __restrict__ mn, const unsigned* __restrict__ mx,
                        const float* __restrict__ deg, float* __restrict__ agg5,
                        int N, int FI) {
  int i = blockIdx.x * blockDim.x + threadIdx.x;
  if (i >= N * FI) return;
  int n = i / FI, c = i - n * FI;
  float cnt = deg[n], cc = fmaxf(cnt, 1.f);
  float sv = s[i], mean = sv / cc;
  float m2 = s2[i] / cc;
  float sd = sqrtf(fmaxf(m2 - mean * mean, 0.f) + 1e-5f);
  bool has = cnt > 0.f;
  float mnv = has ? fdec(mn[i]) : 0.f;
  float mxv = has ? fdec(mx[i]) : 0.f;
  float* o = agg5 + (size_t)n * 5 * FI;
  o[0 * FI + c] = sv;  o[1 * FI + c] = mean; o[2 * FI + c] = mnv;
  o[3 * FI + c] = mxv; o[4 * FI + c] = sd;
}

// one 8-wide run of the virtual [x | aggr*scales] feature row (never straddles
// a region boundary: runs are 8-aligned, regions are FI-aligned, FI%8==0)
template <int FI>
__device__ __forceinline__ void feat8(const float* __restrict__ xin,
                                      const float* __restrict__ agg5,
                                      const float* __restrict__ scl,
                                      int node, int rb, f4v& u, f4v& v) {
  const f4v* p;
  float sc;
  if (rb < FI) {
    p = (const f4v*)&xin[(size_t)node * FI + rb];
    sc = 1.f;
  } else {
    int aa = rb - FI;
    int g = aa / (5 * FI);
    int r = aa - g * (5 * FI);
    p = (const f4v*)&agg5[(size_t)node * 5 * FI + r];
    sc = scl[node * 5 + g];
  }
  u = p[0] * sc;
  v = p[1] * sc;
}

// ---------------------------------------------------------------------------
// fused post GEMM: on-the-fly features -> Wpost -> (LDS bf16 restage) -> Wlin
// ---------------------------------------------------------------------------
template <int FI, int FO>
__global__ void post_gemm(const float* __restrict__ xin, const float* __restrict__ agg5,
                          const float* __restrict__ scl,
                          const __bf16* __restrict__ Ppost, const float* __restrict__ bpost,
                          const __bf16* __restrict__ Plin, const float* __restrict__ blin,
                          float* __restrict__ out, int ldo, int ocol, int N) {
  constexpr int K1 = 26 * FI;     // 1664 / 832
  constexpr int NJ = FO / 16;
  constexpr int NW = 4;
  __shared__ __bf16 ldsT[NW][16 * FO];

  const int w = threadIdx.x >> 5, lane = threadIdx.x & 31;
  const int nb = (blockIdx.x * NW + w) * 16;
  if (nb >= N) return;
  const int half = lane >> 4, mrow = lane & 15, col = lane & 15;
  const int node = nb + mrow;

  v8f acc[NJ] = {};
#pragma unroll 1
  for (int kt = 0; kt < K1 / 32; ++kt) {
    int kb = kt * 32 + half * 8;
    f4v a0, a1, a2, a3;
    feat8<FI>(xin, agg5, scl, node, kb, a0, a1);
    feat8<FI>(xin, agg5, scl, node, kb + 16, a2, a3);
    v16bf av = mk16(a0, a1, a2, a3);
#pragma unroll
    for (int j = 0; j < NJ; ++j)
      acc[j] = wmma_bf16(av, ldB(Ppost, kt, NJ, j, lane), acc[j]);
  }

  // bias, restage as bf16 for the chained Wlin GEMM
#pragma unroll
  for (int j = 0; j < NJ; ++j) {
    float bb = bpost[j * 16 + col];
#pragma unroll
    for (int r = 0; r < 8; ++r)
      ldsT[w][(r + 8 * half) * FO + j * 16 + col] = (__bf16)(acc[j][r] + bb);
  }
  asm volatile("s_wait_dscnt 0x0" ::: "memory");

  v8f acc2[NJ] = {};
#pragma unroll
  for (int kt = 0; kt < FO / 32; ++kt) {
    const v8bf* lo = (const v8bf*)&ldsT[w][mrow * FO + kt * 32 + half * 8];
    const v8bf* hi = (const v8bf*)&ldsT[w][mrow * FO + kt * 32 + 16 + half * 8];
    v16bf av = cat16(lo[0], hi[0]);
#pragma unroll
    for (int j = 0; j < NJ; ++j)
      acc2[j] = wmma_bf16(av, ldB(Plin, kt, NJ, j, lane), acc2[j]);
  }
#pragma unroll
  for (int j = 0; j < NJ; ++j) {
    float bb = blin[j * 16 + col];
#pragma unroll
    for (int r = 0; r < 8; ++r)
      out[(size_t)(nb + r + 8 * half) * ldo + ocol + j * 16 + col] = acc2[j][r] + bb;
  }
}

// ---------------------------------------------------------------------------
// generic node GEMM; A fp32 [N x lda] (lda % 32 == 0, zero-padded), packed B.
// one wave = one 16x16 output tile (j = blockIdx.y of NJ).
// stores cols [0,FOST): value for c<FO, 0 for pad cols. act: 0 none 1 relu 2 sig
// ---------------------------------------------------------------------------
__global__ void node_gemm(const float* __restrict__ A, int lda,
                          const __bf16* __restrict__ P, const float* __restrict__ bias,
                          float* __restrict__ out, int ldo, int ocol,
                          int N, int KT, int NJ, int FO, int FOST, int act) {
  const int w = threadIdx.x >> 5, lane = threadIdx.x & 31;
  const int nb = (blockIdx.x * (blockDim.x >> 5) + w) * 16;
  if (nb >= N) return;
  const int j = blockIdx.y;
  const int half = lane >> 4, mrow = lane & 15, col = lane & 15;
  const float* Arow = A + (size_t)(nb + mrow) * lda;

  v8f acc = {};
#pragma unroll 1
  for (int kt = 0; kt < KT; ++kt) {
    int kb = kt * 32 + half * 8;
    const f4v* p0 = (const f4v*)&Arow[kb];
    const f4v* p1 = (const f4v*)&Arow[kb + 16];
    v16bf av = mk16(p0[0], p0[1], p1[0], p1[1]);
    acc = wmma_bf16(av, ldB(P, kt, NJ, j, lane), acc);
  }
  int c = j * 16 + col;
  if (c < FOST) {
    float bb = (c < FO) ? bias[c] : 0.f;
#pragma unroll
    for (int r = 0; r < 8; ++r) {
      float v = acc[r] + bb;
      if (act == 1) v = fmaxf(v, 0.f);
      else if (act == 2) v = __builtin_amdgcn_rcpf(1.f + __expf(-v));
      if (c >= FO) v = 0.f;   // zero pad columns
      out[(size_t)(nb + r + 8 * half) * ldo + ocol + c] = v;
    }
  }
}

// ---------------------------------------------------------------------------
extern "C" void kernel_launch(void* const* d_in, const int* in_sizes, int n_in,
                              void* d_out, int out_size, void* d_ws, size_t ws_size,
                              hipStream_t stream) {
  (void)in_sizes; (void)n_in; (void)out_size; (void)ws_size;
  const int N = 50000, E = 800000;

  const float* x  = (const float*)d_in[0];
  const int*   ei = (const int*)d_in[1];
  const float *Wpre1 = (const float*)d_in[2],  *bpre1 = (const float*)d_in[3];
  const float *Wpost1= (const float*)d_in[4],  *bpost1= (const float*)d_in[5];
  const float *Wlin1 = (const float*)d_in[6],  *blin1 = (const float*)d_in[7];
  const float *Wpre2 = (const float*)d_in[8],  *bpre2 = (const float*)d_in[9];
  const float *Wpost2= (const float*)d_in[10], *bpost2= (const float*)d_in[11];
  const float *Wlin2 = (const float*)d_in[12], *blin2 = (const float*)d_in[13];
  const float *Wpre3 = (const float*)d_in[14], *bpre3 = (const float*)d_in[15];
  const float *Wpost3= (const float*)d_in[16], *bpost3= (const float*)d_in[17];
  const float *Wlin3 = (const float*)d_in[18], *blin3 = (const float*)d_in[19];
  const float *Wn1 = (const float*)d_in[20], *bn1 = (const float*)d_in[21];
  const float *Wa1 = (const float*)d_in[22], *ba1 = (const float*)d_in[23];
  const float *Wa3 = (const float*)d_in[24], *ba3 = (const float*)d_in[25];
  const float *Wf1 = (const float*)d_in[26], *bf1 = (const float*)d_in[27];
  const float *Wf2 = (const float*)d_in[28], *bf2 = (const float*)d_in[29];
  const float *Wf3 = (const float*)d_in[30], *bf3 = (const float*)d_in[31];

  // workspace carve-out (~206 MB with reuse)
  char* ws = (char*)d_ws;
  size_t off = 0;
  auto take = [&](size_t bytes) -> char* {
    char* p = ws + off;
    off = (off + bytes + 255) & ~(size_t)255;
    return p;
  };
  float*    deg  = (float*)take((size_t)N * 4);
  float*    sums = (float*)take(2 * 4);
  float*    scl  = (float*)take((size_t)N * 5 * 4);
  float*    s    = (float*)take((size_t)N * 64 * 4);
  float*    s2   = (float*)take((size_t)N * 64 * 4);
  unsigned* mn   = (unsigned*)take((size_t)N * 64 * 4);
  unsigned* mx   = (unsigned*)take((size_t)N * 64 * 4);
  float*    agg5 = (float*)take((size_t)N * 320 * 4);
  float*    x1   = (float*)take((size_t)N * 32 * 4);
  float*    x2   = (float*)take((size_t)N * 64 * 4);
  float*    h    = (float*)take((size_t)N * 352 * 4);
  float*    h2   = agg5;   // N*320 exactly; agg5 dead after layer-3 post GEMM
  float*    h3   = s;      // N*128 over s+s2 region (N*128 == N*64 * 2)
  float*    y    = (float*)d_out;

  // packed bf16 weights (fragment order); sizes = KT*NJ*512 bf16 elements
  struct PW { const float* W; int K, FO, KT, NJ; __bf16* P; };
  PW pw[15] = {
    {Wpre1, 128, 64, 4, 4, nullptr},  {Wpre2, 64, 32, 2, 2, nullptr},
    {Wpre3, 128, 64, 4, 4, nullptr},
    {Wpost1, 1664, 32, 52, 2, nullptr}, {Wpost2, 832, 64, 26, 4, nullptr},
    {Wpost3, 1664, 128, 52, 8, nullptr},
    {Wlin1, 32, 32, 1, 2, nullptr},   {Wlin2, 64, 64, 2, 4, nullptr},
    {Wlin3, 128, 128, 4, 8, nullptr},
    {Wn1, 64, 32, 2, 2, nullptr},     {Wa1, 32, 64, 1, 4, nullptr},
    {Wa3, 64, 128, 2, 8, nullptr},
    {Wf1, 352, 300, 11, 20, nullptr}, {Wf2, 300, 124, 10, 8, nullptr},
    {Wf3, 124, 1, 4, 1, nullptr},
  };
  for (int i = 0; i < 15; ++i)
    pw[i].P = (__bf16*)take((size_t)pw[i].KT * pw[i].NJ * 512 * 2);
  for (int i = 0; i < 15; ++i) {
    int total = pw[i].KT * pw[i].NJ * 512;
    pack_w<<<(total + 255) / 256, 256, 0, stream>>>(pw[i].W, pw[i].P,
                                                    pw[i].K, pw[i].FO,
                                                    pw[i].KT, pw[i].NJ);
  }
  __bf16 *pWpre1 = pw[0].P, *pWpre2 = pw[1].P, *pWpre3 = pw[2].P;
  __bf16 *pWpost1 = pw[3].P, *pWpost2 = pw[4].P, *pWpost3 = pw[5].P;
  __bf16 *pWlin1 = pw[6].P, *pWlin2 = pw[7].P, *pWlin3 = pw[8].P;
  __bf16 *pWn1 = pw[9].P, *pWa1 = pw[10].P, *pWa3 = pw[11].P;
  __bf16 *pWf1 = pw[12].P, *pWf2 = pw[13].P, *pWf3 = pw[14].P;

  const int T = 256;
  const int TILES = N / 16;   // 3125, exact
  const int ET    = E / 16;   // 50000, exact

  // degree + PNA scalers
  fill_f32<<<(N + T - 1) / T, T, 0, stream>>>(deg, N, 0.f);
  fill_f32<<<1, 32, 0, stream>>>(sums, 2, 0.f);
  deg_kernel<<<(E + T - 1) / T, T, 0, stream>>>(ei, E, deg);
  reduce_deg<<<(N + 255) / 256, 256, 0, stream>>>(deg, N, sums);
  scl_kernel<<<(N + T - 1) / T, T, 0, stream>>>(deg, sums, scl, N);

  // ---- PNA layer 1: fi=64 -> fo=32, input x, output x1 ----
  agg_init_k<<<(N * 64 + T - 1) / T, T, 0, stream>>>(s, s2, mn, mx, N * 64);
  edge_msg_kernel<64><<<(ET + 7) / 8, 256, 0, stream>>>(x, ei, E, pWpre1, bpre1, s, s2, mn, mx);
  agg_fin<<<(N * 64 + T - 1) / T, T, 0, stream>>>(s, s2, mn, mx, deg, agg5, N, 64);
  post_gemm<64, 32><<<(TILES + 3) / 4, 128, 0, stream>>>(x, agg5, scl, pWpost1, bpost1,
                                                         pWlin1, blin1, x1, 32, 0, N);

  // ---- PNA layer 2: fi=32 -> fo=64, input x1, output x2 ----
  agg_init_k<<<(N * 32 + T - 1) / T, T, 0, stream>>>(s, s2, mn, mx, N * 32);
  edge_msg_kernel<32><<<(ET + 7) / 8, 256, 0, stream>>>(x1, ei, E, pWpre2, bpre2, s, s2, mn, mx);
  agg_fin<<<(N * 32 + T - 1) / T, T, 0, stream>>>(s, s2, mn, mx, deg, agg5, N, 32);
  post_gemm<32, 64><<<(TILES + 3) / 4, 128, 0, stream>>>(x1, agg5, scl, pWpost2, bpost2,
                                                         pWlin2, blin2, x2, 64, 0, N);

  // ---- PNA layer 3: fi=64 -> fo=128, input x2, output h[:,32:160] ----
  agg_init_k<<<(N * 64 + T - 1) / T, T, 0, stream>>>(s, s2, mn, mx, N * 64);
  edge_msg_kernel<64><<<(ET + 7) / 8, 256, 0, stream>>>(x2, ei, E, pWpre3, bpre3, s, s2, mn, mx);
  agg_fin<<<(N * 64 + T - 1) / T, T, 0, stream>>>(s, s2, mn, mx, deg, agg5, N, 64);
  post_gemm<64, 128><<<(TILES + 3) / 4, 128, 0, stream>>>(x2, agg5, scl, pWpost3, bpost3,
                                                          pWlin3, blin3, h, 352, 32, N);

  // ---- skip branches into h = [gg1 | x3 | xa1 | xa2] ----
  const int GX = (TILES + 7) / 8;
  node_gemm<<<dim3(GX, 2), 256, 0, stream>>>(x,  64, pWn1, bn1, h, 352, 0,
                                             N, 2, 2, 32, 32, 1);    // gg1
  node_gemm<<<dim3(GX, 4), 256, 0, stream>>>(x1, 32, pWa1, ba1, h, 352, 160,
                                             N, 1, 4, 64, 64, 1);    // xa1
  node_gemm<<<dim3(GX, 8), 256, 0, stream>>>(x2, 64, pWa3, ba3, h, 352, 224,
                                             N, 2, 8, 128, 128, 1);  // xa2

  // ---- MLP head (h2 ld=320, h3 ld=128, zero-padded) ----
  node_gemm<<<dim3(GX, 20), 256, 0, stream>>>(h,  352, pWf1, bf1, h2, 320, 0,
                                              N, 11, 20, 300, 320, 1);
  node_gemm<<<dim3(GX, 8),  256, 0, stream>>>(h2, 320, pWf2, bf2, h3, 128, 0,
                                              N, 10, 8, 124, 128, 1);
  node_gemm<<<dim3(GX, 1),  256, 0, stream>>>(h3, 128, pWf3, bf3, y,  1,   0,
                                              N, 4, 1, 1, 1, 2);
}
